// Head_53798760350338
// MI455X (gfx1250) — matrix operand order repaired and checked by reading, not compiled, not run on hardware
//
#include <hip/hip_runtime.h>

typedef __attribute__((ext_vector_type(16))) _Float16 v16h;
typedef __attribute__((ext_vector_type(8)))  _Float16 v8h;
typedef __attribute__((ext_vector_type(8)))  float    v8f;

#define EMB_DIM  256
#define HEAD_DIM 32
#define BATCH    4
#define SEQ      4096
#define NSPLIT   4
#define KSPAN    (SEQ / NSPLIT)   // keys per split = 1024

static __device__ __forceinline__ v8f wmma_f16(v16h a, v16h b, v8f c) {
  // D = A(16x32 f16) * B(32x16 f16) + C(16x16 f32)
  return __builtin_amdgcn_wmma_f32_16x16x32_f16(false, a, false, b, (short)0, c,
                                                false, false);
}

// ---------------------------------------------------------------------------
// Kernel 0: transpose the three weight matrices [256][32]f32 into f16
// Wt[3][32][256] so projection B-fragments become contiguous 32B loads.
// ---------------------------------------------------------------------------
__global__ __launch_bounds__(256)
void w_transpose_kernel(const float* __restrict__ Wq,
                        const float* __restrict__ Wk,
                        const float* __restrict__ Wv,
                        _Float16* __restrict__ Wt) {
  const int tid = blockIdx.x * 256 + threadIdx.x;   // 0 .. EMB*HEAD-1
  const int k = tid >> 5;        // row in W   (0..255)
  const int n = tid & 31;        // col in W   (0..31)
  const long o = (long)n * EMB_DIM + k;
  Wt[o]                            = (_Float16)Wq[tid];
  Wt[(long)HEAD_DIM * EMB_DIM + o] = (_Float16)Wk[tid];
  Wt[2L * HEAD_DIM * EMB_DIM + o]  = (_Float16)Wv[tid];
}

// ---------------------------------------------------------------------------
// One K-chunk's worth of raw operands for the projection kernel.
// ---------------------------------------------------------------------------
struct ProjChunk {
  v8f  xa, xb;                          // x slice (fp32, pre-conversion)
  v16h bq0, bq1, bk0, bk1, bv0, bv1;    // W B-fragments
};

static __device__ __forceinline__
ProjChunk load_proj_chunk(const float* __restrict__ x, long arow,
                          const _Float16* __restrict__ Wtq,
                          const _Float16* __restrict__ Wtk,
                          const _Float16* __restrict__ Wtv,
                          int kc, int ln, int hl) {
  ProjChunk c;
  c.xa = *(const v8f*)(x + arow * EMB_DIM + kc + 16 * hl);
  c.xb = *(const v8f*)(x + arow * EMB_DIM + kc + 16 * hl + 8);
  const long o0 = (long)ln * EMB_DIM + kc + 16 * hl;
  const long o1 = (long)(16 + ln) * EMB_DIM + kc + 16 * hl;
  c.bq0 = *(const v16h*)(Wtq + o0);
  c.bq1 = *(const v16h*)(Wtq + o1);
  c.bk0 = *(const v16h*)(Wtk + o0);
  c.bk1 = *(const v16h*)(Wtk + o1);
  c.bv0 = *(const v16h*)(Wtv + o0);
  c.bv1 = *(const v16h*)(Wtv + o1);
  return c;
}

// ---------------------------------------------------------------------------
// Kernel 1: QKV projection.  x[B*S,256]f32 @ W -> f16 Q,K (row major
// [token][32]) and V transposed ([b][32][S]).
// Block: 256 threads = 8 waves; wave w handles 16 tokens.
// Software-pipelined: next chunk's x/W loads issue before the current chunk's
// conversions + 6 WMMAs, so WMMAs don't stall on fresh loads.
// ---------------------------------------------------------------------------
__global__ __launch_bounds__(256)
void qkv_proj_kernel(const float* __restrict__ x,
                     const _Float16* __restrict__ Wt,   // [3][32][256]
                     _Float16* __restrict__ Q,
                     _Float16* __restrict__ K,
                     _Float16* __restrict__ Vt) {
  const int lane = threadIdx.x & 31;
  const int wave = threadIdx.x >> 5;
  const int hl   = lane >> 4;
  const int ln   = lane & 15;
  const long rowBase = (long)blockIdx.x * 128 + wave * 16;
  const long arow    = rowBase + ln;          // A-matrix row M = lane%16

  const _Float16* Wtq = Wt;
  const _Float16* Wtk = Wt + (long)HEAD_DIM * EMB_DIM;
  const _Float16* Wtv = Wt + 2L * HEAD_DIM * EMB_DIM;

  v8f cq0 = {}, cq1 = {}, ck0 = {}, ck1 = {}, cv0 = {}, cv1 = {};

  ProjChunk cur = load_proj_chunk(x, arow, Wtq, Wtk, Wtv, 0, ln, hl);

#pragma unroll
  for (int kc = 0; kc < EMB_DIM; kc += 32) {
    // Issue next chunk's loads NOW (dummy reload of chunk 0 on last iter).
    const int kcn = (kc + 32 < EMB_DIM) ? (kc + 32) : 0;
    ProjChunk nxt = load_proj_chunk(x, arow, Wtq, Wtk, Wtv, kcn, ln, hl);

    // A fragment: a[e] = x[arow][kc + e + 16*hl]  (f32 -> f16)
    v16h a;
#pragma unroll
    for (int e = 0; e < 8; ++e) {
      a[e]     = (_Float16)cur.xa[e];
      a[8 + e] = (_Float16)cur.xb[e];
    }

    cq0 = wmma_f16(a, cur.bq0, cq0);  cq1 = wmma_f16(a, cur.bq1, cq1);
    ck0 = wmma_f16(a, cur.bk0, ck0);  ck1 = wmma_f16(a, cur.bk1, ck1);
    cv0 = wmma_f16(a, cur.bv0, cv0);  cv1 = wmma_f16(a, cur.bv1, cv1);

    cur = nxt;
  }

  // C fragment: lane L, reg r -> (row = rowBase + r + 8*hl, col = ln [+16])
#pragma unroll
  for (int r = 0; r < 8; ++r) {
    const long row = rowBase + r + 8 * hl;
    Q[row * HEAD_DIM + ln]      = (_Float16)cq0[r];
    Q[row * HEAD_DIM + 16 + ln] = (_Float16)cq1[r];
    K[row * HEAD_DIM + ln]      = (_Float16)ck0[r];
    K[row * HEAD_DIM + 16 + ln] = (_Float16)ck1[r];
    const long bidx = row / SEQ, s = row % SEQ;
    _Float16* vtb = Vt + bidx * (long)HEAD_DIM * SEQ;
    vtb[(long)ln * SEQ + s]        = (_Float16)cv0[r];
    vtb[(long)(16 + ln) * SEQ + s] = (_Float16)cv1[r];
  }
}

// ---------------------------------------------------------------------------
// Tile fragment loader for the attention kernel (32 keys).
// ---------------------------------------------------------------------------
static __device__ __forceinline__
void load_kv_tile(const _Float16* __restrict__ Kb,
                  const _Float16* __restrict__ Vb,
                  int kt, int ln, int hl,
                  v16h& ak0, v16h& ak1, v16h& bv0, v16h& bv1) {
  // A fragments: K tile rows (keys).  a[e] = K[key][e+16*hl]
  v8h k0a = *(const v8h*)(Kb + (long)(kt + ln) * HEAD_DIM + 16 * hl);
  v8h k0b = *(const v8h*)(Kb + (long)(kt + ln) * HEAD_DIM + 16 * hl + 8);
  v8h k1a = *(const v8h*)(Kb + (long)(kt + 16 + ln) * HEAD_DIM + 16 * hl);
  v8h k1b = *(const v8h*)(Kb + (long)(kt + 16 + ln) * HEAD_DIM + 16 * hl + 8);
  ak0 = __builtin_shufflevector(k0a, k0b, 0,1,2,3,4,5,6,7,8,9,10,11,12,13,14,15);
  ak1 = __builtin_shufflevector(k1a, k1b, 0,1,2,3,4,5,6,7,8,9,10,11,12,13,14,15);
  // B fragments: V tile from transposed V.  b[e] = V[kt + (e%8) + 16*(e/8) + 8*hl][dim]
  v8h v0a = *(const v8h*)(Vb + (long)ln * SEQ + kt + 8 * hl);
  v8h v0b = *(const v8h*)(Vb + (long)ln * SEQ + kt + 16 + 8 * hl);
  v8h v1a = *(const v8h*)(Vb + (long)(16 + ln) * SEQ + kt + 8 * hl);
  v8h v1b = *(const v8h*)(Vb + (long)(16 + ln) * SEQ + kt + 16 + 8 * hl);
  bv0 = __builtin_shufflevector(v0a, v0b, 0,1,2,3,4,5,6,7,8,9,10,11,12,13,14,15);
  bv1 = __builtin_shufflevector(v1a, v1b, 0,1,2,3,4,5,6,7,8,9,10,11,12,13,14,15);
}

// ---------------------------------------------------------------------------
// Kernel 2: split-K flash attention.  Each wave owns 16 queries and one of
// NSPLIT key ranges (1024 keys = 32 tiles), writing unnormalized partials
// (O_i, m_i, l_i).  Scores are computed TRANSPOSED (E^T = K·Q^T) so the f32
// score fragment maps element-for-element onto the f16 A-fragment of P·V.
// Software-pipelined: next tile's K/V fragments load while the softmax VALU
// work of the current tile executes.
// ---------------------------------------------------------------------------
__global__ __launch_bounds__(256)
void flash_attn_partial_kernel(const _Float16* __restrict__ Q,
                               const _Float16* __restrict__ K,
                               const _Float16* __restrict__ Vt,
                               float* __restrict__ Opart,   // [NSPLIT][B*S][32]
                               float* __restrict__ Mpart,   // [NSPLIT][B*S]
                               float* __restrict__ Lpart) { // [NSPLIT][B*S]
  const int lane = threadIdx.x & 31;
  const int wave = threadIdx.x >> 5;
  const int hl   = lane >> 4;
  const int ln   = lane & 15;
  const int b    = blockIdx.y;
  const int sp   = blockIdx.z;
  const int kb0  = sp * KSPAN;
  const int kend = kb0 + KSPAN;
  const long qBase = (long)blockIdx.x * 128 + wave * 16;   // within sequence
  const _Float16* Qb = Q  + ((long)b * SEQ) * HEAD_DIM;
  const _Float16* Kb = K  + ((long)b * SEQ) * HEAD_DIM;
  const _Float16* Vb = Vt + (long)b * HEAD_DIM * SEQ;

  // B fragment Q^T: b[e] = Q[qBase + ln][e + 16*hl]  (contiguous 32B)
  const v16h bq = *(const v16h*)(Qb + (qBase + ln) * HEAD_DIM + 16 * hl);

  // Per-query online-softmax stats live per LANE (query = ln, replicated in
  // both wave halves).
  float m = -3.0e38f, l = 0.0f;
  v8f oa = {}, ob = {};   // O accumulator: dims 0..15 / 16..31

  // Prologue: fragments for the first tile.
  v16h ak0, ak1, bv0, bv1;
  load_kv_tile(Kb, Vb, kb0, ln, hl, ak0, ak1, bv0, bv1);

  for (int kt = kb0; kt < kend; kt += 32) {
    v8f z = {};
    v8f st0 = wmma_f16(ak0, bq, z);   // E^T keys kt..kt+15   (reg r -> key r+8*hl)
    v8f st1 = wmma_f16(ak1, bq, z);   // E^T keys kt+16..kt+31

    // Issue next tile's loads NOW; the softmax below hides their latency.
    const int ktn = (kt + 32 < kend) ? (kt + 32) : kb0;   // last iter: dummy reload
    v16h nak0, nak1, nbv0, nbv1;
    load_kv_tile(Kb, Vb, ktn, ln, hl, nak0, nak1, nbv0, nbv1);

    // --- online softmax over this 32-key tile (per query = per lane) ---
    float x0 = fmaxf(st0[0], st0[1]), x1 = fmaxf(st0[2], st0[3]);
    float x2 = fmaxf(st0[4], st0[5]), x3 = fmaxf(st0[6], st0[7]);
    float x4 = fmaxf(st1[0], st1[1]), x5 = fmaxf(st1[2], st1[3]);
    float x6 = fmaxf(st1[4], st1[5]), x7 = fmaxf(st1[6], st1[7]);
    float mx = fmaxf(fmaxf(fmaxf(x0, x1), fmaxf(x2, x3)),
                     fmaxf(fmaxf(x4, x5), fmaxf(x6, x7)));
    mx = fmaxf(mx, __shfl_xor(mx, 16, 32));   // join the two wave halves
    const float mnew = fmaxf(m, mx);
    const float cf   = __expf(m - mnew);
    m = mnew;

    // P fragment: the E^T register layout IS the P·V A-fragment layout.
    v16h ap;
    float rs = 0.0f;
#pragma unroll
    for (int r = 0; r < 8; ++r) {
      float p0 = __expf(st0[r] - mnew);
      float p1 = __expf(st1[r] - mnew);
      rs += p0 + p1;
      ap[r]     = (_Float16)p0;
      ap[8 + r] = (_Float16)p1;
    }
    rs += __shfl_xor(rs, 16, 32);
    l = l * cf + rs;

    // Rescale accumulators: O reg r holds query r+8*hl; fetch that query's
    // correction factor from the lane that owns it.
#pragma unroll
    for (int r = 0; r < 8; ++r) {
      float cfq = __shfl(cf, r + 8 * hl, 32);
      oa[r] *= cfq;
      ob[r] *= cfq;
    }

    oa = wmma_f16(ap, bv0, oa);
    ob = wmma_f16(ap, bv1, ob);

    // Rotate the pipeline buffers.
    ak0 = nak0; ak1 = nak1; bv0 = nbv0; bv1 = nbv1;
  }

  // Write unnormalized partials + stats for the combine pass.
  float* op = Opart + ((long)sp * BATCH * SEQ) * HEAD_DIM
            + ((long)b * SEQ) * HEAD_DIM;
#pragma unroll
  for (int r = 0; r < 8; ++r) {
    const long row = qBase + r + 8 * hl;
    op[row * HEAD_DIM + ln]      = oa[r];
    op[row * HEAD_DIM + 16 + ln] = ob[r];
  }
  if (hl == 0) {
    const long t = (long)sp * BATCH * SEQ + (long)b * SEQ + qBase + ln;
    Mpart[t] = m;
    Lpart[t] = l;
  }
}

// ---------------------------------------------------------------------------
// Kernel 3: combine the NSPLIT partials:
//   out = (sum_i e^{m_i-m} O_i) / (sum_i e^{m_i-m} l_i) * d^-0.5
// One thread per output element.
// ---------------------------------------------------------------------------
__global__ __launch_bounds__(256)
void flash_combine_kernel(const float* __restrict__ Opart,
                          const float* __restrict__ Mpart,
                          const float* __restrict__ Lpart,
                          float* __restrict__ out) {
  const long idx = (long)blockIdx.x * 256 + threadIdx.x;  // [0, B*S*32)
  const long t = idx >> 5;                                 // token
  const int  d = (int)(idx & 31);

  float mi[NSPLIT];
  float mt = -3.0e38f;
#pragma unroll
  for (int i = 0; i < NSPLIT; ++i) {
    mi[i] = Mpart[(long)i * BATCH * SEQ + t];
    mt = fmaxf(mt, mi[i]);
  }
  float acc = 0.0f, lsum = 0.0f;
#pragma unroll
  for (int i = 0; i < NSPLIT; ++i) {
    const float w = __expf(mi[i] - mt);
    acc  += w * Opart[((long)i * BATCH * SEQ + t) * HEAD_DIM + d];
    lsum += w * Lpart[(long)i * BATCH * SEQ + t];
  }
  const float scale = 0.17677669529663687f;   // 32^-0.5 (post-softmax, per ref)
  out[t * HEAD_DIM + d] = acc * (scale / lsum);
}

// ---------------------------------------------------------------------------
extern "C" void kernel_launch(void* const* d_in, const int* in_sizes, int n_in,
                              void* d_out, int out_size, void* d_ws, size_t ws_size,
                              hipStream_t stream) {
  const float* x  = (const float*)d_in[0];
  const float* Wq = (const float*)d_in[1];
  const float* Wk = (const float*)d_in[2];
  const float* Wv = (const float*)d_in[3];
  float* out = (float*)d_out;

  const size_t tok   = (size_t)BATCH * SEQ;        // 16384 tokens
  const size_t qkv   = tok * HEAD_DIM;             // elements per f16 matrix
  _Float16* Qh = (_Float16*)d_ws;                  // [B*S][32]        1 MB
  _Float16* Kh = Qh + qkv;                         // [B*S][32]        1 MB
  _Float16* Vt = Kh + qkv;                         // [B][32][S]       1 MB
  _Float16* Wt = Vt + qkv;                         // [3][32][256]     48 KB
  float* Opart = (float*)(Wt + 3L * HEAD_DIM * EMB_DIM);  // [NSPLIT][B*S][32] 8 MB
  float* Mpart = Opart + (size_t)NSPLIT * qkv;            // [NSPLIT][B*S]    256 KB
  float* Lpart = Mpart + (size_t)NSPLIT * tok;            // [NSPLIT][B*S]    256 KB

  w_transpose_kernel<<<dim3((EMB_DIM * HEAD_DIM) / 256), 256, 0, stream>>>(
      Wq, Wk, Wv, Wt);
  qkv_proj_kernel<<<dim3((BATCH * SEQ) / 128), 256, 0, stream>>>(x, Wt, Qh, Kh, Vt);
  flash_attn_partial_kernel<<<dim3(SEQ / 128, BATCH, NSPLIT), 256, 0, stream>>>(
      Qh, Kh, Vt, Opart, Mpart, Lpart);
  flash_combine_kernel<<<dim3((BATCH * SEQ * HEAD_DIM) / 256), 256, 0, stream>>>(
      Opart, Mpart, Lpart, out);
}
